// EGCN_HAgent_74431783240162
// MI455X (gfx1250) — matrix operand order repaired and checked by reading, not compiled, not run on hardware
//
#include <hip/hip_runtime.h>
#include <hip/hip_bf16.h>
#include <stdint.h>

typedef __attribute__((ext_vector_type(16))) _Float16 v16h;
typedef __attribute__((ext_vector_type(8)))  _Float16 v8h;
typedef __attribute__((ext_vector_type(8)))  float    v8f;

#define N_NODES 100000
#define N_EDGES 1600000
#define IN_DIM  64
#define HID     128
#define N_ACT   16
#define TOPK    128

// Gather a 16-bit WMMA A fragment (16x32, [m][k] LDS tile, row stride kd halves):
//   lane<16 holds K{kb+0..7, kb+16..23}; lane>=16 holds K{kb+8..15, kb+24..31}.
// All reads are 16B-aligned contiguous -> ds_load_b128.
__device__ __forceinline__ v16h frag_a(const _Float16* sA, int m16, int kd,
                                       int kb, int lhalf) {
  const v8h lo = *(const v8h*)&sA[m16 * kd + kb + lhalf * 8];
  const v8h hi = *(const v8h*)&sA[m16 * kd + kb + 16 + lhalf * 8];
  return __builtin_shufflevector(lo, hi, 0, 1, 2, 3, 4, 5, 6, 7,
                                 8, 9, 10, 11, 12, 13, 14, 15);
}

// Gather a 16-bit WMMA B fragment from a TRANSPOSED [n][k] LDS tile:
//   lane holds column N=lane%16, K{kb+0..7, kb+16..23} (lane<16) or K{kb+8..15, kb+24..31} (lane>=16).
// two 16B chunks (K and K+16) -> 2x ds_load_b128.
__device__ __forceinline__ v16h frag_b(const _Float16* sWt, int n, int kd,
                                       int kb, int lhalf) {
  const int base = n * kd + kb + lhalf * 8;
  const v8h lo = *(const v8h*)&sWt[base];
  const v8h hi = *(const v8h*)&sWt[base + 16];
  return __builtin_shufflevector(lo, hi, 0, 1, 2, 3, 4, 5, 6, 7,
                                 8, 9, 10, 11, 12, 13, 14, 15);
}

// ---------------------------------------------------------------------------
// fc1: x = relu(inputs @ W1 + b1), also emit f16 copy of x for GEMM2.
// Block = 256 threads = 8 waves; block computes 16 rows x 128 cols.
// Each wave: one 16x16 tile via v_wmma_f32_16x16x32_f16 (K=64 -> 2 WMMAs).
// W1 staged TRANSPOSED ([n][k]) so B fragments are contiguous ds_load_b128.
// ---------------------------------------------------------------------------
__global__ __launch_bounds__(256) void k_fc1(const float* __restrict__ in,
        const float* __restrict__ W1, const float* __restrict__ b1,
        float* __restrict__ x, _Float16* __restrict__ xh) {
  __shared__ _Float16 sWt[HID * IN_DIM];  // [n][k], 16 KB
  __shared__ _Float16 sA[16 * IN_DIM];    // [m][k],  2 KB
  const int t = threadIdx.x;
  const int tile = blockIdx.x;            // 16 rows per tile (100000 % 16 == 0)

  for (int i = t; i < IN_DIM * HID; i += 256) {
    int k = i >> 7;                       // i / HID
    int n = i & (HID - 1);                // i % HID
    sWt[n * IN_DIM + k] = (_Float16)W1[i];  // coalesced global read, LDS scatter
  }
  const float* arow = in + (size_t)tile * 16 * IN_DIM;
  for (int i = t; i < 16 * IN_DIM; i += 256) sA[i] = (_Float16)arow[i];
  __syncthreads();

  const int wave = t >> 5, lane = t & 31;
  const int n0 = wave * 16;
  const int lhalf = (lane >= 16) ? 1 : 0;
  const int m16 = lane & 15;
  v8f c = {};
#pragma unroll
  for (int kb = 0; kb < IN_DIM; kb += 32) {
    v16h a = frag_a(sA, m16, IN_DIM, kb, lhalf);
    v16h b = frag_b(sWt, n0 + m16, IN_DIM, kb, lhalf);
    c = __builtin_amdgcn_wmma_f32_16x16x32_f16(false, a, false, b, (short)0, c,
                                               false, false);
  }
  const int col = n0 + m16;
  const float bias = b1[col];
#pragma unroll
  for (int r = 0; r < 8; ++r) {
    int m = r + (lhalf ? 8 : 0);          // C/D layout: lanes 16-31 -> M+8
    float v = c[r] + bias;
    v = v > 0.f ? v : 0.f;
    size_t off = ((size_t)tile * 16 + m) * HID + col;
    x[off] = v;
    xh[off] = (_Float16)v;
  }
}

// ---------------------------------------------------------------------------
// score = tanh((x . p)/||p||); emit 64-bit sort key = ordered(score) || ~index
// so descending-key order == (score desc, index asc), matching lax.top_k ties.
// One wave per row (row = 128 floats = 512B coalesced).
// ---------------------------------------------------------------------------
__global__ __launch_bounds__(256) void k_score(const float* __restrict__ x,
        const float* __restrict__ p, float* __restrict__ score,
        unsigned long long* __restrict__ keys) {
  const int wave = threadIdx.x >> 5, lane = threadIdx.x & 31;
  const int row = blockIdx.x * 8 + wave;
  if (row >= N_NODES) return;
  const float4 pv = ((const float4*)p)[lane];
  const float4 xv = ((const float4*)(x + (size_t)row * HID))[lane];
  float d  = xv.x * pv.x + xv.y * pv.y + xv.z * pv.z + xv.w * pv.w;
  float pn = pv.x * pv.x + pv.y * pv.y + pv.z * pv.z + pv.w * pv.w;
#pragma unroll
  for (int o = 16; o > 0; o >>= 1) {
    d  += __shfl_down(d, o, 32);
    pn += __shfl_down(pn, o, 32);
  }
  if (lane == 0) {
    float s = tanhf(d * rsqrtf(pn));
    score[row] = s;
    unsigned u = __float_as_uint(s);
    u ^= ((unsigned)((int)u >> 31)) | 0x80000000u;   // order-preserving flip
    keys[row] = ((unsigned long long)u << 32) | (unsigned)(~(unsigned)row);
  }
}

// ---------------------------------------------------------------------------
// Degree (with self loop) -> dinv = rsqrt(deg)
// ---------------------------------------------------------------------------
__global__ void k_deg_init(float* deg) {
  int i = blockIdx.x * 256 + threadIdx.x;
  if (i < N_NODES) deg[i] = 1.0f;                    // self loop
}
__global__ void k_deg_edges(const int* __restrict__ col, float* deg) {
  int e = blockIdx.x * 256 + threadIdx.x;
  if (e < N_EDGES) atomicAdd(&deg[col[e]], 1.0f);
}
__global__ void k_dinv(float* deg) {
  int i = blockIdx.x * 256 + threadIdx.x;
  if (i < N_NODES) deg[i] = rsqrtf(deg[i]);
}

// ---------------------------------------------------------------------------
// Exact top-128: 64-bit bitwise binary search for the 128th-largest key
// (keys are unique: index embedded), then collect + rank-sort 128 items.
// Single block of 1024 threads; keys (800 KB) are L2-resident across passes.
// ---------------------------------------------------------------------------
__global__ __launch_bounds__(1024) void k_topk(
        const unsigned long long* __restrict__ keys,
        const float* __restrict__ score,
        int* __restrict__ perm, float* __restrict__ vals) {
  __shared__ unsigned s_cnt;
  __shared__ unsigned long long s_sel[TOPK];
  __shared__ unsigned s_n;
  const int t = threadIdx.x;
  unsigned long long ans = 0ull;
  for (int bit = 63; bit >= 0; --bit) {
    unsigned long long cand = ans | (1ull << bit);
    if (t == 0) s_cnt = 0;
    __syncthreads();
    unsigned local = 0;
    for (int i = t; i < N_NODES; i += 1024) local += (keys[i] >= cand) ? 1u : 0u;
    if (local) atomicAdd(&s_cnt, local);
    __syncthreads();
    if (s_cnt >= TOPK) ans = cand;                   // uniform across block
    __syncthreads();
  }
  if (t == 0) s_n = 0;
  __syncthreads();
  for (int i = t; i < N_NODES; i += 1024) {
    unsigned long long k = keys[i];
    if (k >= ans) {
      unsigned pos = atomicAdd(&s_n, 1u);
      if (pos < TOPK) s_sel[pos] = k;
    }
  }
  __syncthreads();
  if (t < TOPK) {
    unsigned long long k = s_sel[t];
    int rank = 0;
    for (int j = 0; j < TOPK; ++j) rank += (s_sel[j] > k) ? 1 : 0;
    int idx = (int)(~(unsigned)(k & 0xFFFFFFFFull));
    perm[rank] = idx;
    vals[rank] = score[idx];
  }
}

// x_tilde[k] = x[perm[k]] * vals[k]
__global__ void k_xtilde(const float* __restrict__ x, const int* __restrict__ perm,
        const float* __restrict__ vals, float* __restrict__ xt) {
  int b = blockIdx.x, t = threadIdx.x;
  xt[b * HID + t] = x[(size_t)perm[b] * HID + t] * vals[b];
}

// ---------------------------------------------------------------------------
// GRU step (batch=128, H=128): one block per batch row, one thread per column.
// Emits W_evolved TRANSPOSED as f16 ([n][k]) so GEMM2 can stage it directly.
// ---------------------------------------------------------------------------
__global__ __launch_bounds__(HID) void k_gru(const float* __restrict__ xt,
        const float* __restrict__ h0,
        const float* __restrict__ Wih, const float* __restrict__ Whh,
        const float* __restrict__ bih, const float* __restrict__ bhh,
        _Float16* __restrict__ Wevh_t) {
  __shared__ float sx[HID], sh[HID];
  const int b = blockIdx.x, t = threadIdx.x;
  sx[t] = xt[b * HID + t];
  sh[t] = h0[b * HID + t];
  __syncthreads();
  float ir = bih[t], iz = bih[HID + t], in = bih[2 * HID + t];
  float hr = bhh[t], hz = bhh[HID + t], hn = bhh[2 * HID + t];
  const float* wr = Wih + (size_t)t * HID;
  const float* wz = Wih + (size_t)(HID + t) * HID;
  const float* wn = Wih + (size_t)(2 * HID + t) * HID;
  const float* vr = Whh + (size_t)t * HID;
  const float* vz = Whh + (size_t)(HID + t) * HID;
  const float* vn = Whh + (size_t)(2 * HID + t) * HID;
  for (int k = 0; k < HID; ++k) {
    float xv = sx[k], hv = sh[k];
    ir += xv * wr[k]; iz += xv * wz[k]; in += xv * wn[k];
    hr += hv * vr[k]; hz += hv * vz[k]; hn += hv * vn[k];
  }
  float r = 1.f / (1.f + expf(-(ir + hr)));
  float z = 1.f / (1.f + expf(-(iz + hz)));
  float n = tanhf(in + r * hn);
  float out = (1.f - z) * n + z * sh[t];
  // W_evolved[k=b][n=t], stored transposed: [n][k]
  Wevh_t[(size_t)t * HID + b] = (_Float16)out;
}

// ---------------------------------------------------------------------------
// GEMM2: xw = x_f16 @ W_evolved_f16 (K=128 -> 4 WMMAs per tile), fp32 out.
// W arrives already transposed ([n][k]) -> contiguous B fragments.
// ---------------------------------------------------------------------------
__global__ __launch_bounds__(256) void k_gemm2(const _Float16* __restrict__ xh,
        const _Float16* __restrict__ Wt, float* __restrict__ xw) {
  __shared__ _Float16 sWt[HID * HID];  // 32 KB, [n][k]
  __shared__ _Float16 sA[16 * HID];    //  4 KB, [m][k]
  const int t = threadIdx.x, tile = blockIdx.x;
  for (int i = t; i < HID * HID; i += 256) sWt[i] = Wt[i];
  const _Float16* ar = xh + (size_t)tile * 16 * HID;
  for (int i = t; i < 16 * HID; i += 256) sA[i] = ar[i];
  __syncthreads();

  const int wave = t >> 5, lane = t & 31;
  const int n0 = wave * 16;
  const int lhalf = (lane >= 16) ? 1 : 0;
  const int m16 = lane & 15;
  v8f c = {};
#pragma unroll
  for (int kb = 0; kb < HID; kb += 32) {
    v16h a = frag_a(sA, m16, HID, kb, lhalf);
    v16h b = frag_b(sWt, n0 + m16, HID, kb, lhalf);
    c = __builtin_amdgcn_wmma_f32_16x16x32_f16(false, a, false, b, (short)0, c,
                                               false, false);
  }
  const int col = n0 + m16;
#pragma unroll
  for (int r = 0; r < 8; ++r) {
    int m = r + (lhalf ? 8 : 0);
    xw[((size_t)tile * 16 + m) * HID + col] = c[r];
  }
}

// ---------------------------------------------------------------------------
// h (= d_out h-region, [N,256]): cols 0..127 init w/ self-loop term, 128..255 = x
// ---------------------------------------------------------------------------
__global__ void k_init_h(const float* __restrict__ xw, const float* __restrict__ x,
        const float* __restrict__ dinv, float* __restrict__ h) {
  size_t i = (size_t)blockIdx.x * 256 + threadIdx.x;
  if (i >= (size_t)N_NODES * HID) return;
  size_t n = i / HID, c = i % HID;
  float d = dinv[n];
  h[n * 2 * HID + c]        = d * d * xw[i];
  h[n * 2 * HID + HID + c]  = x[i];
}

// ---------------------------------------------------------------------------
// Edge scatter: h[col] += dinv[row]*dinv[col] * xw[row]. One wave per edge,
// float4 gather per lane, 4 f32 atomics. xw (51MB) + h (102MB) fit in L2.
// Prefetch next edge's source row (global_prefetch_b8).
// ---------------------------------------------------------------------------
__global__ __launch_bounds__(256) void k_scatter(const int* __restrict__ row,
        const int* __restrict__ col, const float* __restrict__ dinv,
        const float* __restrict__ xw, float* __restrict__ h) {
  const int wave = threadIdx.x >> 5, lane = threadIdx.x & 31;
  const int stride = gridDim.x * 8;
  for (int e = blockIdx.x * 8 + wave; e < N_EDGES; e += stride) {
    int e2 = e + stride;
    if (e2 < N_EDGES)
      __builtin_prefetch(&xw[(size_t)row[e2] * HID + lane * 4], 0, 1);
    int r = row[e], c = col[e];
    float w = dinv[r] * dinv[c];
    float4 v = ((const float4*)(xw + (size_t)r * HID))[lane];
    float* dst = h + (size_t)c * 2 * HID + lane * 4;
    atomicAdd(dst + 0, w * v.x);
    atomicAdd(dst + 1, w * v.y);
    atomicAdd(dst + 2, w * v.z);
    atomicAdd(dst + 3, w * v.w);
  }
}

// ---------------------------------------------------------------------------
// fc2: q = h @ W2 + b2. 16 lanes per node share broadcast h loads; W2 in LDS.
// ---------------------------------------------------------------------------
__global__ __launch_bounds__(256) void k_fc2(const float* __restrict__ h,
        const float* __restrict__ W2, const float* __restrict__ b2,
        float* __restrict__ q) {
  __shared__ float sW[2 * HID * N_ACT];  // 16 KB
  for (int i = threadIdx.x; i < 2 * HID * N_ACT; i += 256) sW[i] = W2[i];
  __syncthreads();
  size_t tid = (size_t)blockIdx.x * 256 + threadIdx.x;
  size_t n = tid / N_ACT;
  int a = (int)(tid % N_ACT);
  if (n >= (size_t)N_NODES) return;
  const float* hr = h + n * 2 * HID;
  float acc = b2[a];
#pragma unroll 8
  for (int c = 0; c < 2 * HID; ++c) acc += hr[c] * sW[c * N_ACT + a];
  q[n * N_ACT + a] = acc;
}

// ---------------------------------------------------------------------------
extern "C" void kernel_launch(void* const* d_in, const int* in_sizes, int n_in,
                              void* d_out, int out_size, void* d_ws, size_t ws_size,
                              hipStream_t stream) {
  (void)in_sizes; (void)n_in; (void)out_size; (void)ws_size;
  const float* inputs = (const float*)d_in[0];
  /* d_in[1] hidden_states: unused by the reference body */
  const int*   ei  = (const int*)d_in[2];
  const float* W1  = (const float*)d_in[3];
  const float* b1  = (const float*)d_in[4];
  const float* p   = (const float*)d_in[5];
  const float* Wih = (const float*)d_in[6];
  const float* Whh = (const float*)d_in[7];
  const float* bih = (const float*)d_in[8];
  const float* bhh = (const float*)d_in[9];
  const float* iw  = (const float*)d_in[10];   // initial_weight [1,128,128]
  const float* W2  = (const float*)d_in[11];
  const float* b2  = (const float*)d_in[12];

  char* ws = (char*)d_ws;
  size_t o = 0;
  auto alloc = [&](size_t bytes) -> void* {
    void* ptr = ws + o;
    o = (o + bytes + 255) & ~(size_t)255;
    return ptr;
  };
  float*              x     = (float*)alloc((size_t)N_NODES * HID * 4);
  _Float16*           xh    = (_Float16*)alloc((size_t)N_NODES * HID * 2);
  float*              xw    = (float*)alloc((size_t)N_NODES * HID * 4);
  float*              score = (float*)alloc((size_t)N_NODES * 4);
  unsigned long long* keys  = (unsigned long long*)alloc((size_t)N_NODES * 8);
  float*              dinv  = (float*)alloc((size_t)N_NODES * 4);
  _Float16*           Wevh  = (_Float16*)alloc((size_t)HID * HID * 2);  // [n][k]
  float*              xt    = (float*)alloc((size_t)HID * HID * 4);
  int*                perm  = (int*)alloc(TOPK * 4);
  float*              vals  = (float*)alloc(TOPK * 4);

  float* qout = (float*)d_out;                       // [N, 16]
  float* hout = qout + (size_t)N_NODES * N_ACT;      // [N, 256]

  const int* erow = ei;                // edge_index[0]
  const int* ecol = ei + N_EDGES;      // edge_index[1]

  k_fc1   <<<N_NODES / 16, 256, 0, stream>>>(inputs, W1, b1, x, xh);
  k_score <<<(N_NODES + 7) / 8, 256, 0, stream>>>(x, p, score, keys);
  k_deg_init <<<(N_NODES + 255) / 256, 256, 0, stream>>>(dinv);
  k_deg_edges<<<(N_EDGES + 255) / 256, 256, 0, stream>>>(ecol, dinv);
  k_dinv  <<<(N_NODES + 255) / 256, 256, 0, stream>>>(dinv);
  k_topk  <<<1, 1024, 0, stream>>>(keys, score, perm, vals);
  k_xtilde<<<TOPK, HID, 0, stream>>>(x, perm, vals, xt);
  k_gru   <<<HID, HID, 0, stream>>>(xt, iw, Wih, Whh, bih, bhh, Wevh);
  k_gemm2 <<<N_NODES / 16, 256, 0, stream>>>(xh, Wevh, xw);
  k_init_h<<<(int)(((size_t)N_NODES * HID + 255) / 256), 256, 0, stream>>>(xw, x, dinv, hout);
  k_scatter<<<8192, 256, 0, stream>>>(erow, ecol, dinv, xw, hout);
  k_fc2   <<<(int)(((size_t)N_NODES * N_ACT + 255) / 256), 256, 0, stream>>>(hout, W2, b2, qout);
}
